// HopfieldMemoryBank_87522843558250
// MI455X (gfx1250) — matrix operand order repaired and checked by reading, not compiled, not run on hardware
//
#include <hip/hip_runtime.h>

#define D_MODEL 1024
#define NMEM 64
#define MTOT 32768   // 8 * 4096 rows
#define NUM_STEPS 3

typedef __attribute__((ext_vector_type(16))) __bf16       v16bf;
typedef __attribute__((ext_vector_type(8)))  float        v8f;
typedef __attribute__((ext_vector_type(4)))  unsigned int v4u;
typedef __attribute__((ext_vector_type(4)))  int          v4i;
typedef __attribute__((ext_vector_type(4)))  float        v4f;

union FragBF16 {
  v4u   q[2];   // 32 bytes = 16 bf16
  v16bf v;
};

// ---- CDNA5 async global->LDS path (guarded; falls back to sync copy) ------
#if __has_builtin(__builtin_amdgcn_global_load_async_to_lds_b128) && \
    __has_builtin(__builtin_amdgcn_s_wait_asynccnt)
#define USE_ASYNC_LDS 1
#else
#define USE_ASYNC_LDS 0
#endif

#if USE_ASYNC_LDS
typedef __attribute__((address_space(1))) v4i* gv4i_p;   // global int4*
typedef __attribute__((address_space(3))) v4i* lv4i_p;   // LDS int4*
#endif

__device__ __forceinline__ void async_ld16(unsigned short* lds_dst,
                                           const unsigned short* gsrc) {
#if USE_ASYNC_LDS
  __builtin_amdgcn_global_load_async_to_lds_b128((gv4i_p)gsrc, (lv4i_p)lds_dst, 0, 0);
#else
  *(v4u*)lds_dst = *(const v4u*)gsrc;
#endif
}
__device__ __forceinline__ void async_wait_all() {
#if USE_ASYNC_LDS
  __builtin_amdgcn_s_wait_asynccnt(0);
#endif
}

__device__ __forceinline__ unsigned short f2bf(float f) {
  union { float f; unsigned int u; } c; c.f = f;
  unsigned int u = c.u;
  u += 0x7FFFu + ((u >> 16) & 1u);   // round-to-nearest-even
  return (unsigned short)(u >> 16);
}
__device__ __forceinline__ float bf2f(unsigned short h) {
  union { unsigned int u; float f; } c; c.u = ((unsigned int)h) << 16;
  return c.f;
}
__device__ __forceinline__ v8f wmma_bf16(const FragBF16& a, const FragBF16& b, v8f c) {
  return __builtin_amdgcn_wmma_f32_16x16x32_bf16(false, a.v, false, b.v,
                                                 (short)0, c, false, false);
}

// ---------------------------------------------------------------------------
// Kernel 1: convert weights to bf16 (transposed) + memory to bf16 (both layouts)
// ---------------------------------------------------------------------------
__global__ __launch_bounds__(256) void prep_kernel(
    const float* __restrict__ Wq, const float* __restrict__ Wo,
    const float* __restrict__ mem,
    unsigned short* __restrict__ WqT, unsigned short* __restrict__ WoT,
    unsigned short* __restrict__ memRM, unsigned short* __restrict__ memT) {
  const int WN = D_MODEL * D_MODEL;
  int idx = blockIdx.x * blockDim.x + threadIdx.x;
  if (idx < WN) {
    int d = idx >> 10, e = idx & 1023;                 // Wq[d][e] -> WqT[e][d]
    WqT[(size_t)e * D_MODEL + d] = f2bf(Wq[idx]);
  } else if (idx < 2 * WN) {
    int i = idx - WN;
    int d = i >> 10, e = i & 1023;
    WoT[(size_t)e * D_MODEL + d] = f2bf(Wo[i]);
  } else if (idx < 2 * WN + NMEM * D_MODEL) {
    int i = idx - 2 * WN;
    int n = i >> 10, d = i & 1023;                     // mem[n][d]
    unsigned short v = f2bf(mem[i]);
    memRM[i] = v;
    memT[(size_t)d * NMEM + n] = v;
  }
}

// ---------------------------------------------------------------------------
// Kernel 2/4: Out_bf16[m][n] = bf16( A[m][:] @ W[:, n] + bias[n] )
// W given transposed (WT[n][k], bf16). A is fp32 (x) or bf16 (xi).
// Tiling: 128x128 block, BK=64, 8 waves (4x2), each wave 32x64 (2x4 WMMA tiles).
// Panels staged in LDS via async global->LDS copies (when available).
// ---------------------------------------------------------------------------
template <bool A_F32>
__global__ __launch_bounds__(256) void gemm_bf16_kernel(
    const void* __restrict__ Aptr, const unsigned short* __restrict__ WT,
    const float* __restrict__ bias, unsigned short* __restrict__ Out) {
  __shared__ unsigned short As[128][72];   // 144B row stride: conflict-free
  __shared__ unsigned short Bs[128][72];   // stored as [out-col][k]

  const int tid   = threadIdx.x;
  const int lane  = tid & 31;
  const int wave  = tid >> 5;
  const int laneL = lane & 15;
  const int laneH = lane >> 4;
  const int waveM = wave >> 1;   // 0..3
  const int waveN = wave & 1;    // 0..1
  const int m0 = blockIdx.y * 128;
  const int n0 = blockIdx.x * 128;

  v8f acc[2][4];
  for (int i = 0; i < 2; i++)
    for (int j = 0; j < 4; j++)
      for (int k = 0; k < 8; k++) acc[i][j][k] = 0.0f;

  const int lr = tid >> 1;   // 0..127: panel row loaded by this thread
  const int lh = tid & 1;    // which 32-element half of the row

  for (int kk = 0; kk < D_MODEL; kk += 64) {
    // ---- stage A panel (128 x 64) into LDS as bf16 ----
    if (A_F32) {
      const float* A = (const float*)Aptr;
      const float* src = A + (size_t)(m0 + lr) * D_MODEL + kk + lh * 32;
      if (kk + 64 < D_MODEL) __builtin_prefetch(src + 64, 0, 0);
      float fv[32];
      for (int i = 0; i < 8; i++) {
        v4f f = ((const v4f*)src)[i];
        fv[4 * i] = f[0]; fv[4 * i + 1] = f[1];
        fv[4 * i + 2] = f[2]; fv[4 * i + 3] = f[3];
      }
      for (int c = 0; c < 4; c++) {
        v4u q;
        for (int i = 0; i < 4; i++)
          q[i] = (unsigned)f2bf(fv[8 * c + 2 * i]) |
                 ((unsigned)f2bf(fv[8 * c + 2 * i + 1]) << 16);
        *(v4u*)&As[lr][lh * 32 + 8 * c] = q;
      }
    } else {
      const unsigned short* A = (const unsigned short*)Aptr;
      const unsigned short* src = A + (size_t)(m0 + lr) * D_MODEL + kk + lh * 32;
      for (int c = 0; c < 4; c++)
        async_ld16(&As[lr][lh * 32 + 8 * c], src + 8 * c);
    }
    // ---- stage B panel (128 out-cols x 64 k) from WT (bf16, row = out-col) ----
    {
      const unsigned short* src = WT + (size_t)(n0 + lr) * D_MODEL + kk + lh * 32;
      for (int c = 0; c < 4; c++)
        async_ld16(&Bs[lr][lh * 32 + 8 * c], src + 8 * c);
    }
    async_wait_all();
    __syncthreads();

    // ---- 2 K-steps x 8 WMMAs per wave ----
    for (int ks = 0; ks < 2; ks++) {
      const int kb = ks * 32;
      FragBF16 a[2], b[4];
      for (int mt = 0; mt < 2; mt++) {
        int row = waveM * 32 + mt * 16 + laneL;
        // A layout: lanes 0-15: K{0..7,16..23}; lanes 16-31: K{8..15,24..31}
        a[mt].q[0] = *(const v4u*)&As[row][kb + laneH * 8];
        a[mt].q[1] = *(const v4u*)&As[row][kb + laneH * 8 + 16];
      }
      for (int nt = 0; nt < 4; nt++) {
        int col = waveN * 64 + nt * 16 + laneL;
        // B layout: lanes 0-15: K{0..7,16..23}; lanes 16-31: K{8..15,24..31}
        b[nt].q[0] = *(const v4u*)&Bs[col][kb + laneH * 8];
        b[nt].q[1] = *(const v4u*)&Bs[col][kb + laneH * 8 + 16];
      }
      for (int mt = 0; mt < 2; mt++)
        for (int nt = 0; nt < 4; nt++)
          acc[mt][nt] = wmma_bf16(a[mt], b[nt], acc[mt][nt]);
    }
    __syncthreads();
  }

  // ---- epilogue: bias + bf16 store. C layout: row = v + 8*laneH, col = laneL ----
  for (int nt = 0; nt < 4; nt++) {
    int col = n0 + waveN * 64 + nt * 16 + laneL;
    float bv = bias[col];
    for (int mt = 0; mt < 2; mt++) {
      int rbase = m0 + waveM * 32 + mt * 16 + laneH * 8;
      for (int v = 0; v < 8; v++)
        Out[(size_t)(rbase + v) * D_MODEL + col] = f2bf(acc[mt][nt][v] + bv);
    }
  }
}

// ---------------------------------------------------------------------------
// Kernel 3: fused 3-step Hopfield iteration. One wave per 16-row tile; the
// xi tile lives in LDS across all steps (no HBM round-trips between steps).
// ---------------------------------------------------------------------------
__global__ __launch_bounds__(32) void hopfield_kernel(
    const unsigned short* __restrict__ memRM,  // [64][1024] bf16
    const unsigned short* __restrict__ memT,   // [1024][64] bf16
    const float* __restrict__ log_beta,
    unsigned short* __restrict__ xi) {         // [32768][1024] bf16, in/out
  __shared__ unsigned short xiS[16][1032];     // 2064B row stride (16B-mult, conflict-free)
  __shared__ unsigned short attnS[16][72];     // 144B row stride

  const int lane  = threadIdx.x;
  const int laneL = lane & 15;
  const int laneH = lane >> 4;
  const size_t r0 = (size_t)blockIdx.x * 16;
  const float beta = __expf(log_beta[0]);

  // load xi tile: each lane copies half a row (512 bf16 = 64 x 16B), async to LDS
  {
    int row = lane >> 1, half = lane & 1;
    const unsigned short* src = xi + (r0 + row) * D_MODEL + half * 512;
    unsigned short* dst = &xiS[row][half * 512];
    for (int i = 0; i < 64; i++) async_ld16(dst + 8 * i, src + 8 * i);
    async_wait_all();
  }
  __syncthreads();

  for (int step = 0; step < NUM_STEPS; step++) {
    // ---- scores = xi @ mem^T : 16 x 64, K = 1024 ----
    v8f s[4];
    for (int nt = 0; nt < 4; nt++)
      for (int v = 0; v < 8; v++) s[nt][v] = 0.0f;

    for (int kt = 0; kt < 32; kt++) {
      const int kb = kt * 32;
      FragBF16 a;
      a.q[0] = *(const v4u*)&xiS[laneL][kb + laneH * 8];
      a.q[1] = *(const v4u*)&xiS[laneL][kb + laneH * 8 + 16];
      for (int nt = 0; nt < 4; nt++) {
        FragBF16 b;
        const unsigned short* brow =
            memRM + (size_t)(nt * 16 + laneL) * D_MODEL + kb + laneH * 8;
        b.q[0] = ((const v4u*)brow)[0];
        b.q[1] = ((const v4u*)brow)[2];
        s[nt] = wmma_bf16(a, b, s[nt]);
      }
    }

    // ---- softmax over 64 cols. Row r = v + 8*laneH lives in the 16-lane half;
    //      reduce across the 4 tile registers + shfl_xor within the half. ----
    float mx[8], rs[8];
    for (int v = 0; v < 8; v++) {
      float m = fmaxf(fmaxf(s[0][v], s[1][v]), fmaxf(s[2][v], s[3][v]));
      for (int off = 1; off < 16; off <<= 1) m = fmaxf(m, __shfl_xor(m, off, 32));
      mx[v] = m;
    }
    for (int v = 0; v < 8; v++) {
      float sum = 0.0f;
      for (int nt = 0; nt < 4; nt++) {
        float p = __expf(beta * (s[nt][v] - mx[v]));
        s[nt][v] = p;
        sum += p;
      }
      for (int off = 1; off < 16; off <<= 1) sum += __shfl_xor(sum, off, 32);
      rs[v] = 1.0f / sum;
    }
    // stage attn (bf16) into LDS in row-major form for A-fragment loads
    for (int nt = 0; nt < 4; nt++)
      for (int v = 0; v < 8; v++)
        attnS[v + laneH * 8][nt * 16 + laneL] = f2bf(s[nt][v] * rs[v]);

    // ---- xi = attn @ mem : 16 x 1024, K = 64 (two WMMAs per 16-col tile) ----
    for (int et = 0; et < 64; et++) {
      v8f u;
      for (int v = 0; v < 8; v++) u[v] = 0.0f;
      for (int ks = 0; ks < 2; ks++) {
        const int nb = ks * 32;
        FragBF16 a, b;
        a.q[0] = *(const v4u*)&attnS[laneL][nb + laneH * 8];
        a.q[1] = *(const v4u*)&attnS[laneL][nb + laneH * 8 + 16];
        const unsigned short* brow =
            memT + (size_t)(et * 16 + laneL) * NMEM + nb + laneH * 8;
        b.q[0] = ((const v4u*)brow)[0];
        b.q[1] = ((const v4u*)brow)[2];
        u = wmma_bf16(a, b, u);
      }
      for (int v = 0; v < 8; v++)
        xiS[v + laneH * 8][et * 16 + laneL] = f2bf(u[v]);
    }
  }

  // write back xi tile (sync path: compiler orders vs. prior ds stores)
  {
    int row = lane >> 1, half = lane & 1;
    v4u* dst = (v4u*)(xi + (r0 + row) * D_MODEL + half * 512);
    const v4u* src = (const v4u*)&xiS[row][half * 512];
    for (int i = 0; i < 64; i++) dst[i] = src[i];
  }
}

// ---------------------------------------------------------------------------
// Kernel 5: y = LayerNorm(x + out2) ; one row per block, 256 threads x 4 cols
// ---------------------------------------------------------------------------
__global__ __launch_bounds__(256) void resid_ln_kernel(
    const float* __restrict__ x, const unsigned short* __restrict__ out2,
    const float* __restrict__ gamma, const float* __restrict__ betap,
    float* __restrict__ y) {
  __shared__ float red[16];
  const int tid = threadIdx.x;
  const size_t row = blockIdx.x;
  const int c = tid * 4;

  v4f xv = *(const v4f*)(x + row * D_MODEL + c);
  const unsigned short* o = out2 + row * D_MODEL + c;
  unsigned long long ov = *(const unsigned long long*)o;
  float h0 = xv[0] + bf2f((unsigned short)(ov & 0xFFFF));
  float h1 = xv[1] + bf2f((unsigned short)((ov >> 16) & 0xFFFF));
  float h2 = xv[2] + bf2f((unsigned short)((ov >> 32) & 0xFFFF));
  float h3 = xv[3] + bf2f((unsigned short)(ov >> 48));

  float s = h0 + h1 + h2 + h3;
  float q = h0 * h0 + h1 * h1 + h2 * h2 + h3 * h3;
  for (int off = 1; off < 32; off <<= 1) {
    s += __shfl_xor(s, off, 32);
    q += __shfl_xor(q, off, 32);
  }
  const int wave = tid >> 5, lane = tid & 31;
  if (lane == 0) { red[wave] = s; red[8 + wave] = q; }
  __syncthreads();
  if (tid < 32) {
    float ss = (lane < 8) ? red[lane] : 0.0f;
    float qq = (lane < 8) ? red[8 + lane] : 0.0f;
    for (int off = 1; off < 8; off <<= 1) {
      ss += __shfl_xor(ss, off, 32);
      qq += __shfl_xor(qq, off, 32);
    }
    if (lane == 0) { red[0] = ss; red[1] = qq; }
  }
  __syncthreads();
  const float mean = red[0] * (1.0f / D_MODEL);
  const float var  = red[1] * (1.0f / D_MODEL) - mean * mean;
  const float rstd = rsqrtf(var + 1e-5f);

  v4f g = *(const v4f*)(gamma + c);
  v4f b = *(const v4f*)(betap + c);
  v4f out;
  out[0] = (h0 - mean) * rstd * g[0] + b[0];
  out[1] = (h1 - mean) * rstd * g[1] + b[1];
  out[2] = (h2 - mean) * rstd * g[2] + b[2];
  out[3] = (h3 - mean) * rstd * g[3] + b[3];
  *(v4f*)(y + row * D_MODEL + c) = out;
}

// ---------------------------------------------------------------------------
// Launch: prep -> gemm(Wq) -> hopfield(3 steps fused) -> gemm(Wo) -> resid+LN
// ws layout: WqT(2MB) | WoT(2MB) | memRM(128KB) | memT(128KB) | xi(64MB) | out2(64MB)
// ---------------------------------------------------------------------------
extern "C" void kernel_launch(void* const* d_in, const int* in_sizes, int n_in,
                              void* d_out, int out_size, void* d_ws, size_t ws_size,
                              hipStream_t stream) {
  const float* x        = (const float*)d_in[0];
  const float* memory   = (const float*)d_in[1];
  const float* Wq       = (const float*)d_in[2];
  const float* bq       = (const float*)d_in[3];
  const float* Wo       = (const float*)d_in[4];
  const float* bo       = (const float*)d_in[5];
  const float* ln_gamma = (const float*)d_in[6];
  const float* ln_beta  = (const float*)d_in[7];
  const float* log_beta = (const float*)d_in[8];
  float* y = (float*)d_out;

  char* ws = (char*)d_ws;
  unsigned short* WqT   = (unsigned short*)(ws);
  unsigned short* WoT   = (unsigned short*)(ws + (size_t)2 * 1024 * 1024);
  unsigned short* memRM = (unsigned short*)(ws + (size_t)4 * 1024 * 1024);
  unsigned short* memT  = (unsigned short*)(ws + (size_t)4 * 1024 * 1024 + 128 * 1024);
  unsigned short* xibuf = (unsigned short*)(ws + (size_t)4 * 1024 * 1024 + 256 * 1024);
  unsigned short* out2  = (unsigned short*)(ws + (size_t)4 * 1024 * 1024 + 256 * 1024 +
                                            (size_t)MTOT * D_MODEL * 2);

  {  // prep: 2*1M weight elements + 64K memory elements
    int total = 2 * D_MODEL * D_MODEL + NMEM * D_MODEL;
    prep_kernel<<<(total + 255) / 256, 256, 0, stream>>>(Wq, Wo, memory, WqT, WoT,
                                                         memRM, memT);
  }
  // xi = bf16(x @ Wq + bq)
  gemm_bf16_kernel<true><<<dim3(D_MODEL / 128, MTOT / 128), 256, 0, stream>>>(
      (const void*)x, WqT, bq, xibuf);
  // 3 fused Hopfield steps, xi updated in place
  hopfield_kernel<<<MTOT / 16, 32, 0, stream>>>(memRM, memT, log_beta, xibuf);
  // out2 = bf16(xi @ Wo + bo)
  gemm_bf16_kernel<false><<<dim3(D_MODEL / 128, MTOT / 128), 256, 0, stream>>>(
      (const void*)xibuf, WoT, bo, out2);
  // y = LayerNorm(x + out2)
  resid_ln_kernel<<<MTOT, 256, 0, stream>>>(x, out2, ln_gamma, ln_beta, y);
}
